// LSTMLayer_12128987644432
// MI455X (gfx1250) — matrix operand order repaired and checked by reading, not compiled, not run on hardware
//
#include <hip/hip_runtime.h>

// ---------------------------------------------------------------------------
// Types for CDNA5 WMMA (wave32): bf16 A/B fragments (8 VGPRs), f32 C/D (8 VGPRs)
// ---------------------------------------------------------------------------
typedef __bf16 bf16_t;
typedef __attribute__((ext_vector_type(8)))  __bf16 v8bf;
typedef __attribute__((ext_vector_type(16))) __bf16 v16bf;
typedef __attribute__((ext_vector_type(8)))  float  v8f;
typedef int v4i __attribute__((__vector_size__(16)));  // matches async-builtin param

// Problem constants
#define TT   512
#define BB   128
#define HH   512
#define GG   2048            // 4*H
#define KK   512             // I == H
#define NWG  64              // persistent grid: 64 blocks x 4 waves = 256 wave-tiles

// LDS layout: rows padded 1024B -> 1040B so row-strided 16-lane fragment reads
// hit distinct banks (1040/4 = 260 = 4 mod 64 -> rows spread across banks).
#define PADB   1040
#define WSLAB  ((unsigned)(64 * PADB))      // one W slice: 64 rows x 512 K bf16
#define HSLAB  ((unsigned)(4 * 16 * PADB))  // 4 waves x 16 rows x 512 K bf16

// ---------------------------------------------------------------------------
// CDNA5 async global->LDS copy (ASYNCcnt), with sync fallback.
// ---------------------------------------------------------------------------
#if __has_builtin(__builtin_amdgcn_global_load_async_to_lds_b128)
#define HAS_ASYNC 1
#else
#define HAS_ASYNC 0
#endif

__device__ __forceinline__ void async_copy16(const void* g, void* l) {
#if HAS_ASYNC
  __builtin_amdgcn_global_load_async_to_lds_b128(
      (__attribute__((address_space(1))) v4i*)g,
      (__attribute__((address_space(3))) v4i*)l, 0, 0);
#else
  *(v8bf*)l = *(const v8bf*)g;
#endif
}

#if HAS_ASYNC
#if __has_builtin(__builtin_amdgcn_s_wait_asynccnt)
#define WAIT_ASYNC(n) __builtin_amdgcn_s_wait_asynccnt(n)
#else
#define WAIT_ASYNC(n) asm volatile("s_wait_asynccnt %0" ::"i"(n) : "memory")
#endif
#else
// fallback copies are same-wave LDS stores: hardware keeps DS in order, just
// stop the compiler from reordering the aliasing ds_load above the ds_store.
#define WAIT_ASYNC(n) asm volatile("" ::: "memory")
#endif

// ---------------------------------------------------------------------------
// Global-direct fragment loaders (phase-1 GEMM only).
// ---------------------------------------------------------------------------
__device__ __forceinline__ v16bf load_a_frag_g(const bf16_t* __restrict__ base,
                                               int lane, int k0) {
  const int m  = lane & 15;
  const int h8 = (lane >> 4) * 8;
  const bf16_t* p = base + (size_t)m * KK + k0 + h8;
  v8bf lo = *(const v8bf*)p;
  v8bf hi = *(const v8bf*)(p + 16);
  return __builtin_shufflevector(lo, hi, 0,1,2,3,4,5,6,7,8,9,10,11,12,13,14,15);
}

__device__ __forceinline__ v16bf load_b_frag_g(const bf16_t* __restrict__ W,
                                               int gcol0, int lane, int k0) {
  const int n   = lane & 15;
  const int h16 = (lane >> 4) * 8;
  const bf16_t* p = W + (size_t)(gcol0 + n) * KK + k0 + h16;
  v8bf lo = *(const v8bf*)p;
  v8bf hi = *(const v8bf*)(p + 16);
  return __builtin_shufflevector(lo, hi, 0,1,2,3,4,5,6,7,8,9,10,11,12,13,14,15);
}

// ---------------------------------------------------------------------------
// LDS fragment loaders (ds_load_b128 x2 each, bank-conflict-free via PADB).
// ---------------------------------------------------------------------------
__device__ __forceinline__ v16bf lds_a_frag(const char* sl, int lane, int k0) {
  const int m  = lane & 15;
  const int h8 = (lane >> 4) * 8;
  const char* p = sl + m * PADB + (k0 + h8) * 2;
  v8bf lo = *(const v8bf*)p;
  v8bf hi = *(const v8bf*)(p + 32);
  return __builtin_shufflevector(lo, hi, 0,1,2,3,4,5,6,7,8,9,10,11,12,13,14,15);
}

__device__ __forceinline__ v16bf lds_b_frag(const char* wsl, int j, int lane, int k0) {
  const int n   = lane & 15;
  const int h16 = (lane >> 4) * 8;
  const char* p = wsl + (j * 16 + n) * PADB + (k0 + h16) * 2;
  v8bf lo = *(const v8bf*)p;
  v8bf hi = *(const v8bf*)(p + 32);
  return __builtin_shufflevector(lo, hi, 0,1,2,3,4,5,6,7,8,9,10,11,12,13,14,15);
}

__device__ __forceinline__ v8f wmma_bf16(v16bf a, v16bf b, v8f c) {
  return __builtin_amdgcn_wmma_f32_16x16x32_bf16(
      false, a, false, b, (short)0, c, false, false);
}

// ---------------------------------------------------------------------------
// Cooperative W-slice preload: 64 rows (4 gates x 16 cols) x 512 K -> LDS.
// ---------------------------------------------------------------------------
__device__ __forceinline__ void preload_w(char* dst, const bf16_t* __restrict__ W,
                                          int n0, int tid) {
#pragma unroll
  for (int it = 0; it < 32; ++it) {
    const int p   = tid + it * 128;   // 4096 16B pieces
    const int row = p >> 6;           // 0..63  (j*16 + n)
    const int off = p & 63;           // 16B piece within row
    const int j = row >> 4, n = row & 15;
    const bf16_t* g = W + (size_t)(j * HH + n0 + n) * KK + off * 8;
    async_copy16(g, dst + (size_t)row * PADB + off * 16);
  }
}

// Per-wave async copy of 16 activation rows (its m-block) into its LDS slab.
// Issued in K-chunk order: chunk c = async instructions 2c, 2c+1.
__device__ __forceinline__ void copy_rows_async(char* sl,
                                                const bf16_t* __restrict__ src,
                                                int lane) {
#pragma unroll
  for (int c = 0; c < 16; ++c) {
#pragma unroll
    for (int s = 0; s < 2; ++s) {
      const int row   = lane & 15;
      const int piece = (lane >> 4) + 2 * s;
      const bf16_t* g = src + (size_t)row * KK + c * 32 + piece * 8;
      async_copy16(g, sl + row * PADB + c * 64 + piece * 16);
    }
  }
}

// ---------------------------------------------------------------------------
// Quad-tile GEMM out of LDS with incremental ASYNCcnt waits.  Template
// recursion makes every wait count an integer-constant expression.
// Assumes exactly 32 async copies outstanding on entry (chunk c = 2 copies).
// ---------------------------------------------------------------------------
template <int I>
__device__ __forceinline__ void gemm_lds_steps(const char* sl, const char* wsl,
                                               int lane,
                                               v8f& a0, v8f& a1, v8f& a2, v8f& a3) {
  if constexpr (I < 16) {
    WAIT_ASYNC(30 - 2 * I);  // chunk I arrived (async loads complete in order)
    const int k0 = I * 32;
    v16bf aC = lds_a_frag(sl, lane, k0);
    v16bf b0 = lds_b_frag(wsl, 0, lane, k0);
    v16bf b1 = lds_b_frag(wsl, 1, lane, k0);
    v16bf b2 = lds_b_frag(wsl, 2, lane, k0);
    v16bf b3 = lds_b_frag(wsl, 3, lane, k0);
    a0 = wmma_bf16(aC, b0, a0);
    a1 = wmma_bf16(aC, b1, a1);
    a2 = wmma_bf16(aC, b2, a2);
    a3 = wmma_bf16(aC, b3, a3);
    gemm_lds_steps<I + 1>(sl, wsl, lane, a0, a1, a2, a3);
  }
}

__device__ __forceinline__ void gemm_quad_lds(const char* sl, const char* wsl,
                                              int lane,
                                              v8f& a0, v8f& a1, v8f& a2, v8f& a3) {
  gemm_lds_steps<0>(sl, wsl, lane, a0, a1, a2, a3);
}

// Global-direct pipelined version (phase-1 GEMM: occupancy hides latency).
__device__ __forceinline__ void gemm_quad_g(const bf16_t* __restrict__ A,
                                            const bf16_t* __restrict__ W,
                                            int n0, int dn, int lane,
                                            v8f& a0, v8f& a1, v8f& a2, v8f& a3) {
  v16bf aP[2];
  v16bf bP[2][4];
#pragma unroll
  for (int i = 0; i < 2; ++i) {
    const int k0 = i * 32;
    aP[i]    = load_a_frag_g(A, lane, k0);
    bP[i][0] = load_b_frag_g(W, n0 + 0 * dn, lane, k0);
    bP[i][1] = load_b_frag_g(W, n0 + 1 * dn, lane, k0);
    bP[i][2] = load_b_frag_g(W, n0 + 2 * dn, lane, k0);
    bP[i][3] = load_b_frag_g(W, n0 + 3 * dn, lane, k0);
  }
#pragma unroll
  for (int i = 0; i < 16; ++i) {
    const int cur = i & 1;
    a0 = wmma_bf16(aP[cur], bP[cur][0], a0);
    a1 = wmma_bf16(aP[cur], bP[cur][1], a1);
    a2 = wmma_bf16(aP[cur], bP[cur][2], a2);
    a3 = wmma_bf16(aP[cur], bP[cur][3], a3);
    if (i + 2 < 16) {
      const int k0 = (i + 2) * 32;
      aP[cur]    = load_a_frag_g(A, lane, k0);
      bP[cur][0] = load_b_frag_g(W, n0 + 0 * dn, lane, k0);
      bP[cur][1] = load_b_frag_g(W, n0 + 1 * dn, lane, k0);
      bP[cur][2] = load_b_frag_g(W, n0 + 2 * dn, lane, k0);
      bP[cur][3] = load_b_frag_g(W, n0 + 3 * dn, lane, k0);
    }
  }
}

__device__ __forceinline__ float sigf(float x) {
  return 1.0f / (1.0f + __expf(-x));
}

// ---------------------------------------------------------------------------
// Wave-level producer/consumer sync on per-mblk monotonic counters.
// ---------------------------------------------------------------------------
__device__ __forceinline__ void wave_arrive(unsigned* c) {
  __builtin_amdgcn_fence(__ATOMIC_RELEASE, "agent");  // flush this wave's stores
  if ((threadIdx.x & 31) == 0)
    __hip_atomic_fetch_add(c, 1u, __ATOMIC_RELAXED, __HIP_MEMORY_SCOPE_AGENT);
}

__device__ __forceinline__ void wave_wait(unsigned* c, unsigned target) {
  if ((threadIdx.x & 31) == 0) {
    while (__hip_atomic_load(c, __ATOMIC_RELAXED, __HIP_MEMORY_SCOPE_AGENT) < target)
      __builtin_amdgcn_s_sleep(1);
  }
  __builtin_amdgcn_fence(__ATOMIC_ACQUIRE, "agent");  // invalidate stale lines
}

// ---------------------------------------------------------------------------
// Phase 1 (needs big workspace): x_gates[T*B, 4H] = x_bf16 @ W_ih^T (f32 out).
// ---------------------------------------------------------------------------
__global__ __launch_bounds__(256)
void xgate_gemm_kernel(const bf16_t* __restrict__ xbf,
                       const bf16_t* __restrict__ Wih,
                       float* __restrict__ xg) {
  const int lane = threadIdx.x & 31;
  const int wave = threadIdx.x >> 5;
  const int tile = blockIdx.x * 8 + wave;   // 0 .. 131071
  const int mb   = tile >> 5;               // row block of 16 in T*B
  const int g0   = (tile & 31) * 64;        // gate col block of 64
  v8f a0 = {}, a1 = {}, a2 = {}, a3 = {};
  gemm_quad_g(xbf + (size_t)mb * 16 * KK, Wih, g0, 16, lane, a0, a1, a2, a3);
  const int nn = lane & 15;
  const int mh = (lane >> 4) * 8;
#pragma unroll
  for (int r = 0; r < 8; ++r) {
    float* row = xg + (size_t)(mb * 16 + mh + r) * GG + g0 + nn;
    row[0]  = a0[r];
    row[16] = a1[r];
    row[32] = a2[r];
    row[48] = a3[r];
  }
}

// ---------------------------------------------------------------------------
// Persistent recurrence kernel: one launch runs all 512 timesteps.
//  - W slices live in LDS for the whole sequence (loaded once)
//  - h slabs streamed into LDS per step with async copies + incremental waits
//  - cell state c in registers for the whole sequence
//  - per-mblk wave-level sync, h triple-buffered
// ---------------------------------------------------------------------------
template <bool USE_XG>
__global__ __launch_bounds__(128)
void lstm_persistent_kernel(bf16_t* __restrict__ hbuf0,  // h triple buffer (bf16)
                            bf16_t* __restrict__ hbuf1,
                            bf16_t* __restrict__ hbuf2,
                            const bf16_t* __restrict__ xbf,
                            const bf16_t* __restrict__ Wih,
                            const bf16_t* __restrict__ Whh,
                            const float*  __restrict__ xg,
                            const float*  __restrict__ b_ih,
                            const float*  __restrict__ b_hh,
                            const float*  __restrict__ c0,
                            float*  __restrict__ out,        // [T,B,H]
                            float*  __restrict__ hT,
                            float*  __restrict__ cT,
                            unsigned* __restrict__ ctrs) {   // 8 counters, stride 64
  extern __shared__ char smem[];
  const int lane = threadIdx.x & 31;
  const int wave = threadIdx.x >> 5;
  const int tile = blockIdx.x * 4 + wave;   // all 4 waves share hblk
  const int mblk = tile & 7;
  const int m0   = mblk * 16;               // batch rows
  const int n0   = (tile >> 3) * 16;        // h columns

  char* wsl_hh = smem;                       // 64 x PADB
  char* hsl    = smem + WSLAB + wave * 16 * PADB;
  char* wsl_ih = smem + WSLAB + HSLAB;       // fused path only

  // Load weight slices into LDS once; reused for all 512 steps.
  preload_w(wsl_hh, Whh, n0, threadIdx.x);
  if (!USE_XG) preload_w(wsl_ih, Wih, n0, threadIdx.x);
  WAIT_ASYNC(0);
  __syncthreads();

  const int nn  = lane & 15;
  const int mh  = (lane >> 4) * 8;
  const int col = n0 + nn;

  const float bi = b_ih[col]          + b_hh[col];
  const float bf = b_ih[col + HH]     + b_hh[col + HH];
  const float bg = b_ih[col + 2 * HH] + b_hh[col + 2 * HH];
  const float bo = b_ih[col + 3 * HH] + b_hh[col + 3 * HH];

  v8f creg;
#pragma unroll
  for (int r = 0; r < 8; ++r)
    creg[r] = c0[(size_t)(m0 + mh + r) * HH + col];

  bf16_t* hb[3] = {hbuf0, hbuf1, hbuf2};
  unsigned* ctr = ctrs + mblk * 64;

  for (int t = 0; t < TT; ++t) {
    const bf16_t* hprev = hb[t % 3];
    bf16_t* hnext = hb[(t + 1) % 3];
    const float* xg_t = xg + (size_t)t * BB * GG;

    v8f gi = {}, gf = {}, gg = {}, go = {};

    if (USE_XG) {
      // Prefetch this step's x-gate rows so they stream in under the GEMM.
#pragma unroll
      for (int r = 0; r < 8; ++r) {
        const float* xr = xg_t + (size_t)(m0 + mh + r) * GG;
        __builtin_prefetch(xr + col, 0, 0);
        __builtin_prefetch(xr + col + HH, 0, 0);
        __builtin_prefetch(xr + col + 2 * HH, 0, 0);
        __builtin_prefetch(xr + col + 3 * HH, 0, 0);
      }
    } else {
      // Input projection has no cross-wave dependency: overlap it with the
      // producers of h(t) that we have not yet waited for.
      copy_rows_async(hsl, xbf + ((size_t)t * BB + m0) * KK, lane);
      gemm_quad_lds(hsl, wsl_ih, lane, gi, gf, gg, go);
    }

    // Wait for the 32 producers of our 16 batch rows of h(t).
    wave_wait(ctr, 32u * (unsigned)t);

    copy_rows_async(hsl, hprev + (size_t)m0 * KK, lane);
    gemm_quad_lds(hsl, wsl_hh, lane, gi, gf, gg, go);

    float* out_t = out + (size_t)t * BB * HH;
#pragma unroll
    for (int r = 0; r < 8; ++r) {
      const int m = m0 + mh + r;
      float vi = gi[r] + bi;
      float vf = gf[r] + bf;
      float vg = gg[r] + bg;
      float vo = go[r] + bo;
      if (USE_XG) {
        const float* xr = xg_t + (size_t)m * GG;
        vi += xr[col];
        vf += xr[col + HH];
        vg += xr[col + 2 * HH];
        vo += xr[col + 3 * HH];
      }
      const float ii = sigf(vi);
      const float ff = sigf(vf);
      const float G  = tanhf(vg);
      const float oo = sigf(vo);
      const float cn = ff * creg[r] + ii * G;
      const float hn = oo * tanhf(cn);
      creg[r] = cn;
      const size_t ci = (size_t)m * HH + col;
      out_t[ci] = hn;
      hnext[ci] = (bf16_t)hn;
      if (t == TT - 1) {
        hT[ci] = hn;
        cT[ci] = cn;
      }
    }

    wave_arrive(ctr);  // release h(t+1) for our mblk
  }
}

// ---------------------------------------------------------------------------
// Small utility kernels
// ---------------------------------------------------------------------------
__global__ void cvt_f32_bf16_kernel(const float* __restrict__ s,
                                    bf16_t* __restrict__ d, int n) {
  int i = blockIdx.x * blockDim.x + threadIdx.x;
  if (i < n) d[i] = (bf16_t)s[i];
}

__global__ void init_ctr_kernel(unsigned* c) {
  int i = blockIdx.x * blockDim.x + threadIdx.x;
  if (i < 512) c[i] = 0u;
}

// ---------------------------------------------------------------------------
// Workspace layout (bytes, 256-aligned):
//   [0]           W_ih bf16    2,097,152
//   [2,097,152]   W_hh bf16    2,097,152
//   [4,194,304]   h buf 0/1/2  3 x 131,072 (bf16)
//   [4,587,520]   counters     2,048
//   [4,718,592]   x bf16       67,108,864
//   [71,827,456]  x_gates f32  536,870,912  (optional)
// ---------------------------------------------------------------------------
#define OFF_WIH  ((size_t)0)
#define OFF_WHH  ((size_t)2097152)
#define OFF_H0   ((size_t)4194304)
#define OFF_H1   ((size_t)4325376)
#define OFF_H2   ((size_t)4456448)
#define OFF_CTR  ((size_t)4587520)
#define OFF_XBF  ((size_t)4718592)
#define OFF_XG   ((size_t)71827456)
#define WS_FULL  (OFF_XG + (size_t)TT * BB * GG * 4)

extern "C" void kernel_launch(void* const* d_in, const int* in_sizes, int n_in,
                              void* d_out, int out_size, void* d_ws, size_t ws_size,
                              hipStream_t stream) {
  const float* x    = (const float*)d_in[0];
  const float* h0   = (const float*)d_in[1];
  const float* c0   = (const float*)d_in[2];
  const float* Wih  = (const float*)d_in[3];
  const float* Whh  = (const float*)d_in[4];
  const float* bih  = (const float*)d_in[5];
  const float* bhh  = (const float*)d_in[6];
  float* out = (float*)d_out;                  // [T,B,H] ++ hT[B,H] ++ cT[B,H]

  char* ws = (char*)d_ws;
  bf16_t*   Wih_bf = (bf16_t*)(ws + OFF_WIH);
  bf16_t*   Whh_bf = (bf16_t*)(ws + OFF_WHH);
  bf16_t*   hbuf0  = (bf16_t*)(ws + OFF_H0);
  bf16_t*   hbuf1  = (bf16_t*)(ws + OFF_H1);
  bf16_t*   hbuf2  = (bf16_t*)(ws + OFF_H2);
  unsigned* ctrs   = (unsigned*)(ws + OFF_CTR);
  bf16_t*   x_bf   = (bf16_t*)(ws + OFF_XBF);
  float*    xg     = (float*)(ws + OFF_XG);

  const bool use_xg = (ws_size >= WS_FULL);

  const int NW = GG * KK;        // 1,048,576 weight elements per matrix
  const int NX = TT * BB * KK;   // 33,554,432 input elements
  const int NS = BB * HH;        // 65,536 state elements

  cvt_f32_bf16_kernel<<<(NW + 255) / 256, 256, 0, stream>>>(Wih, Wih_bf, NW);
  cvt_f32_bf16_kernel<<<(NW + 255) / 256, 256, 0, stream>>>(Whh, Whh_bf, NW);
  cvt_f32_bf16_kernel<<<(NX + 255) / 256, 256, 0, stream>>>(x, x_bf, NX);
  cvt_f32_bf16_kernel<<<(NS + 255) / 256, 256, 0, stream>>>(h0, hbuf0, NS);
  init_ctr_kernel<<<2, 256, 0, stream>>>(ctrs);

  if (use_xg) {
    xgate_gemm_kernel<<<16384, 256, 0, stream>>>(x_bf, Wih_bf, xg);
  }

  float* hT = out + (size_t)TT * NS;
  float* cT = hT + NS;
  if (use_xg) {
    lstm_persistent_kernel<true><<<NWG, 128, WSLAB + HSLAB, stream>>>(
        hbuf0, hbuf1, hbuf2, x_bf, Wih_bf, Whh_bf, xg, bih, bhh, c0,
        out, hT, cT, ctrs);
  } else {
    lstm_persistent_kernel<false><<<NWG, 128, 2 * WSLAB + HSLAB, stream>>>(
        hbuf0, hbuf1, hbuf2, x_bf, Wih_bf, Whh_bf, nullptr, bih, bhh, c0,
        out, hT, cT, ctrs);
  }
}
